// RGCNconv_37864431682368
// MI455X (gfx1250) — matrix operand order repaired and checked by reading, not compiled, not run on hardware
//
#include <hip/hip_runtime.h>
#include <hip/hip_bf16.h>
#include <math.h>

#define Bn 32
#define Kk 4
#define Nn 512
#define Ff 128

typedef __attribute__((ext_vector_type(16))) __bf16 v16bf;
typedef __attribute__((ext_vector_type(8)))  float  v8f;

static __device__ __forceinline__ v16bf pack16(float4 a, float4 b, float4 c, float4 d) {
  v16bf r;
  r[0]=(__bf16)a.x;  r[1]=(__bf16)a.y;  r[2]=(__bf16)a.z;  r[3]=(__bf16)a.w;
  r[4]=(__bf16)b.x;  r[5]=(__bf16)b.y;  r[6]=(__bf16)b.z;  r[7]=(__bf16)b.w;
  r[8]=(__bf16)c.x;  r[9]=(__bf16)c.y;  r[10]=(__bf16)c.z; r[11]=(__bf16)c.w;
  r[12]=(__bf16)d.x; r[13]=(__bf16)d.y; r[14]=(__bf16)d.z; r[15]=(__bf16)d.w;
  return r;
}

static __device__ __forceinline__ v8f wmma_bf16(v16bf a, v16bf b, v8f c) {
  return __builtin_amdgcn_wmma_f32_16x16x32_bf16(false, a, false, b, (short)0, c, false, false);
}

// ---------------------------------------------------------------------------
// Kernel D: Dsum[b,j] = K + sum_{k,i} E[b,k,i,j]   (coalesced: lane runs over j)
// ---------------------------------------------------------------------------
__global__ __launch_bounds__(256) void kD_colsum(const float* __restrict__ E,
                                                 float* __restrict__ Dsum) {
  int t = blockIdx.x * 256 + threadIdx.x;   // t = b*Nn + j
  int b = t >> 9;
  int j = t & (Nn - 1);
  const float* Eb = E + (size_t)b * Kk * Nn * Nn + j;
  float s = 0.f;
#pragma unroll 1
  for (int k = 0; k < Kk; ++k) {
    const float* p = Eb + (size_t)k * Nn * Nn;
#pragma unroll 8
    for (int i = 0; i < Nn; ++i) s += p[(size_t)i * Nn];
  }
  Dsum[t] = s + (float)Kk;
}

// ---------------------------------------------------------------------------
// Kernel 1: G_t[b,k,f,j] = sum_m RW[m,f,k]*H[b,j,m]   (bf16, f-major -> B-frag ready)
//           S[b,j,l]     = sum_m H[b,j,m]*SW[m,l]     (fp32)
// grid (jt=4, b=32), 256 threads = 8 waves
// ---------------------------------------------------------------------------
__global__ __launch_bounds__(256) void k1_gemm_GS(const float* __restrict__ H,
                                                  const float* __restrict__ RW,
                                                  const float* __restrict__ SW,
                                                  __bf16* __restrict__ Gt,
                                                  float* __restrict__ S) {
  const int jt   = blockIdx.x;
  const int b    = blockIdx.y;
  const int wave = threadIdx.x >> 5;
  const int lane = threadIdx.x & 31;
  const int lhalf = lane >> 4;
  const int l15   = lane & 15;
  const int jbase = jt * 128;
  const float* Hb = H + (size_t)b * Nn * Ff;

  // ---- Part A: G_t = RW^T @ H^T (M = f rows, N = j cols, K = m) ----
  const int f0 = wave * 16;
#pragma unroll 1
  for (int k = 0; k < Kk; ++k) {
    v8f acc[8] = {};
#pragma unroll
    for (int ms = 0; ms < 4; ++ms) {
      const int m0 = ms * 32;
      v16bf afr;
      {
        const float* rp = RW + (size_t)((m0 + lhalf * 8) * Ff + f0 + l15) * Kk + k;
#pragma unroll
        for (int tt = 0; tt < 8; ++tt) {
          afr[tt]     = (__bf16)rp[(size_t)tt * Ff * Kk];
          afr[8 + tt] = (__bf16)rp[(size_t)(tt + 16) * Ff * Kk];
        }
      }
#pragma unroll
      for (int jb = 0; jb < 8; ++jb) {
        const int j = jbase + jb * 16 + l15;
        const float4* hp4 = (const float4*)(Hb + (size_t)j * Ff + m0 + lhalf * 8);
        v16bf bfr = pack16(hp4[0], hp4[1], hp4[4], hp4[5]);
        acc[jb] = wmma_bf16(afr, bfr, acc[jb]);
      }
    }
    __bf16* gb = Gt + (size_t)(b * Kk + k) * Ff * Nn;
#pragma unroll
    for (int jb = 0; jb < 8; ++jb) {
      const int j = jbase + jb * 16 + l15;
#pragma unroll
      for (int e = 0; e < 8; ++e) {
        const int f = f0 + e + lhalf * 8;
        gb[(size_t)f * Nn + j] = (__bf16)acc[jb][e];
      }
    }
  }

  // ---- Part B: S = H @ SW (M = j rows, N = l cols, K = m) ----
  {
    const int j0 = jbase + wave * 16;
    v8f acc[8] = {};
#pragma unroll
    for (int ms = 0; ms < 4; ++ms) {
      const int m0 = ms * 32;
      const float4* hp4 = (const float4*)(Hb + (size_t)(j0 + l15) * Ff + m0 + lhalf * 16);
      v16bf afr = pack16(hp4[0], hp4[1], hp4[2], hp4[3]);
#pragma unroll
      for (int fb = 0; fb < 8; ++fb) {
        const float* sp = SW + (size_t)(m0 + lhalf * 16) * Ff + fb * 16 + l15;
        v16bf bfr;
#pragma unroll
        for (int tt = 0; tt < 16; ++tt)
          bfr[tt] = (__bf16)sp[(size_t)tt * Ff];
        acc[fb] = wmma_bf16(afr, bfr, acc[fb]);
      }
    }
    float* Sb = S + (size_t)b * Nn * Ff;
#pragma unroll
    for (int fb = 0; fb < 8; ++fb) {
      const int l = fb * 16 + l15;
#pragma unroll
      for (int e = 0; e < 8; ++e) {
        const int j = j0 + e + lhalf * 8;
        Sb[(size_t)j * Ff + l] = acc[fb][e];
      }
    }
  }
}

// ---------------------------------------------------------------------------
// Kernel 2: out[b,i,f] = sigmoid( (sum_k E[b,k] @ G[b,k]) / Dsum + S )
// bf16 WMMA, fp32 accum; G tiles staged to LDS via async copy, double-buffered
// grid (it=4, b=32), 256 threads = 8 waves; wave owns 16 i-rows x 128 f
// ---------------------------------------------------------------------------
__global__ __launch_bounds__(256) void k2_gemm_out(const float* __restrict__ E,
                                                   const __bf16* __restrict__ Gt,
                                                   const float* __restrict__ S,
                                                   const float* __restrict__ Dsum,
                                                   float* __restrict__ out) {
  __shared__ __align__(16) char smem[2][8192];   // 128 f-rows x 64 B, double buffered
  const int it   = blockIdx.x;
  const int b    = blockIdx.y;
  const int tid  = threadIdx.x;
  const int wave = tid >> 5;
  const int lane = tid & 31;
  const int lhalf = lane >> 4;
  const int l15   = lane & 15;
  const int i0 = it * 128 + wave * 16;

  // stage one (k, js) G-tile (128 f x 32 j bf16 = 8 KB) into LDS buffer `buf`:
  // 512 chunks of 16 B; thread handles chunks tid and tid+256.
  auto stage = [&](int step, int buf) {
    const int k  = step >> 4;
    const int js = step & 15;
    const __bf16* gb = Gt + (size_t)(b * Kk + k) * Ff * Nn + js * 32;
#pragma unroll
    for (int r = 0; r < 2; ++r) {
      const int c = tid + r * 256;
      const int f = c >> 2;
      const int q = c & 3;
      const __bf16* src = gb + (size_t)f * Nn + q * 8;
      unsigned laddr = (unsigned)(uintptr_t)&smem[buf][f * 64 + q * 16];
      asm volatile("global_load_async_to_lds_b128 %0, %1, off"
                   :: "v"(laddr), "v"(src) : "memory");
    }
  };

  v8f acc[8] = {};
  stage(0, 0);

  int step = 0;
#pragma unroll 1
  for (int k = 0; k < Kk; ++k) {
    const float* Ebk = E + (size_t)(b * Kk + k) * Nn * Nn;
#pragma unroll 1
    for (int js = 0; js < 16; ++js, ++step) {
      // my stage(step) loads done, then everyone's:
      asm volatile("s_wait_asynccnt 0x0" ::: "memory");
      __syncthreads();
      // overlap: issue stage(step+1) into the other buffer
      if (step + 1 < Kk * 16) stage(step + 1, (step + 1) & 1);

      // A frag from E (fp32 -> bf16): row i0+l15, K runs [j0+lhalf*16..+15]
      const float* ep = Ebk + (size_t)(i0 + l15) * Nn + js * 32 + lhalf * 16;
      if (js < 15) __builtin_prefetch(ep + 32, 0, 0);
      const float4* ep4 = (const float4*)ep;
      v16bf afr = pack16(ep4[0], ep4[1], ep4[2], ep4[3]);

      const char* sb = smem[step & 1];
#pragma unroll
      for (int fb = 0; fb < 8; ++fb) {
        const v16bf* bp = (const v16bf*)(sb + (fb * 16 + l15) * 64 + lhalf * 32);
        acc[fb] = wmma_bf16(afr, *bp, acc[fb]);
      }
    }
  }

  // fused epilogue: out = sigmoid(acc / Dsum + S)
  const float* Sb = S + (size_t)b * Nn * Ff;
  const float* Db = Dsum + (size_t)b * Nn;
  float* ob = out + (size_t)b * Nn * Ff;
  float dinv[8];
#pragma unroll
  for (int e = 0; e < 8; ++e) dinv[e] = 1.0f / Db[i0 + e + lhalf * 8];
#pragma unroll
  for (int fb = 0; fb < 8; ++fb) {
    const int f = fb * 16 + l15;
#pragma unroll
    for (int e = 0; e < 8; ++e) {
      const int i = i0 + e + lhalf * 8;
      const float x = acc[fb][e] * dinv[e] + Sb[(size_t)i * Ff + f];
      ob[(size_t)i * Ff + f] = 1.0f / (1.0f + __expf(-x));
    }
  }
}

// ---------------------------------------------------------------------------
extern "C" void kernel_launch(void* const* d_in, const int* in_sizes, int n_in,
                              void* d_out, int out_size, void* d_ws, size_t ws_size,
                              hipStream_t stream) {
  const float* H  = (const float*)d_in[0];   // [B,N,F]
  const float* E  = (const float*)d_in[1];   // [B,K,N,N]
  const float* RW = (const float*)d_in[2];   // [F,F,K]
  const float* SW = (const float*)d_in[3];   // [F,F]
  float* out = (float*)d_out;                // [B,N,F]

  char* ws = (char*)d_ws;
  __bf16* Gt = (__bf16*)ws;                            // 16 MB  [B,K,F,N] bf16
  float*  S  = (float*)(ws + ((size_t)16 << 20));      //  8 MB  [B,N,F]
  float*  Ds = (float*)(ws + ((size_t)24 << 20));      // 64 KB  [B,N]

  kD_colsum  <<<dim3((Bn * Nn) / 256), dim3(256), 0, stream>>>(E, Ds);
  k1_gemm_GS <<<dim3(Nn / 128, Bn),    dim3(256), 0, stream>>>(H, RW, SW, Gt, S);
  k2_gemm_out<<<dim3(Nn / 128, Bn),    dim3(256), 0, stream>>>(E, Gt, S, Ds, out);
}
